// c2vRNNModel_74990128988308
// MI455X (gfx1250) — matrix operand order repaired and checked by reading, not compiled, not run on hardware
//
#include <hip/hip_runtime.h>
#include <hip/hip_bf16.h>

typedef __attribute__((ext_vector_type(16))) __bf16 v16bf;
typedef __attribute__((ext_vector_type(8)))  __bf16 v8bf;
typedef __attribute__((ext_vector_type(4)))  __bf16 v4bf;
typedef __attribute__((ext_vector_type(8)))  float  v8f;

#define BB 16
#define TT 200
#define PP 100
#define EMB 100
#define DD 300
#define QQ 50
#define SKILL 30
#define OUT 50
#define KPAD 320
#define NT 19   // ceil(300/16) N tiles
#define KT 10   // ceil(300/32) K tiles (K padded to 320)
#define NPAD (NT * 16)        // 304
#define ROWTILES 20000        // B*T*P/16
#define WAVES 4
#define WB_ELEMS (NT * KT * 32 * 16)          // 97280 bf16
#define WB_FLOATS ((WB_ELEMS * 2 + 255) / 256 * 64)  // float offset of padded vecs

// Branchless tanh: 1 - 2/(exp(2x)+1).  exp->inf => 1, exp->0 => -1; no NaN,
// no range-split branch (single v_exp_f32 TRANS op, co-executes with WMMA).
__device__ __forceinline__ float fast_tanh(float x) {
    float s = __expf(2.0f * x);
    return 1.0f - 2.0f / (s + 1.0f);
}

// ---------------------------------------------------------------------------
// Kernel 0: pack W_trans (f32, [300,300] row-major, [K][N]) into bf16
// B-fragments laid out exactly as V_WMMA_F32_16X16X32_BF16 wants them:
//   fragment (nt,kt), lane l, element e  ->  W[K = kt*32 + (l>>4)*8 + e + (e>=8 ? 8 : 0)]
//                                             [N = nt*16 + (l&15)]
// One contiguous 32B chunk per lane -> one b128 pair load at the use site.
// Also writes zero-padded b_trans / W_att (NPAD entries) so the score-kernel
// epilogue needs no bounds guard (keeps EXEC full next to the WMMAs).
// ---------------------------------------------------------------------------
__global__ void pack_wtrans_kernel(const float* __restrict__ W,
                                   const float* __restrict__ b_trans,
                                   const float* __restrict__ w_att,
                                   __bf16* __restrict__ wb,
                                   float* __restrict__ bpad,
                                   float* __restrict__ wpad) {
    int g = blockIdx.x * blockDim.x + threadIdx.x;   // 97280 fragment elements
    if (g < NPAD) {
        bpad[g] = (g < DD) ? b_trans[g] : 0.0f;
        wpad[g] = (g < DD) ? w_att[g]   : 0.0f;
    }
    if (g >= WB_ELEMS) return;
    int e    = g & 15;
    int lane = (g >> 4) & 31;
    int kt   = (g >> 9) % KT;
    int nt   = g / (KT * 512);
    int N = nt * 16 + (lane & 15);
    int K = kt * 32 + (lane >> 4) * 8 + e + ((e >> 3) << 3);
    float v = (K < DD && N < DD) ? W[K * DD + N] : 0.0f;
    wb[g] = (__bf16)v;
}

// ---------------------------------------------------------------------------
// Kernel 1: fused  score = tanh(full_embed @ W_trans + b_trans) @ W_att + b_att
// One wave per 16-row tile; bf16 WMMA, f32 accumulation.
// ---------------------------------------------------------------------------
__global__ __launch_bounds__(WAVES * 32)
void score_kernel(const int* __restrict__ idx,
                  const float* __restrict__ nodes,
                  const float* __restrict__ paths,
                  const __bf16* __restrict__ wb,
                  const float* __restrict__ bpad,   // zero-padded b_trans[NPAD]
                  const float* __restrict__ wpad,   // zero-padded W_att[NPAD]
                  const float* __restrict__ b_att,
                  float* __restrict__ scores)   // [B*T*P], written into attn slot
{
    __shared__ __bf16 As[WAVES][16][KPAD];   // 4 * 16 * 320 * 2B = 40 KB

    const int lane    = threadIdx.x & 31;
    const int w       = threadIdx.x >> 5;
    const int rowTile = blockIdx.x * WAVES + w;
    const bool live   = (rowTile < ROWTILES);   // wave-uniform

    if (live) {
        // Gather 16 rows x [start_e | end_e | path_e] -> bf16 in LDS.
        // Vectorized: float4 global loads, 4x-packed bf16 (b64) LDS stores.
        for (int j = lane; j < 16 * (EMB / 4); j += 32) {
            int r  = j / (EMB / 4);
            int c4 = (j % (EMB / 4)) * 4;
            int row = rowTile * 16 + r;
            int i0 = idx[row * 3 + 0];
            int i1 = idx[row * 3 + 1];
            int i2 = idx[row * 3 + 2];
            float4 s  = *(const float4*)(nodes + i0 * EMB + c4);   // start_e
            float4 en = *(const float4*)(nodes + i2 * EMB + c4);   // end_e
            float4 pa = *(const float4*)(paths + i1 * EMB + c4);   // path_e
            v4bf vs = { (__bf16)s.x,  (__bf16)s.y,  (__bf16)s.z,  (__bf16)s.w };
            v4bf ve = { (__bf16)en.x, (__bf16)en.y, (__bf16)en.z, (__bf16)en.w };
            v4bf vp = { (__bf16)pa.x, (__bf16)pa.y, (__bf16)pa.z, (__bf16)pa.w };
            *(v4bf*)&As[w][r][c4]           = vs;
            *(v4bf*)&As[w][r][EMB + c4]     = ve;
            *(v4bf*)&As[w][r][2 * EMB + c4] = vp;
        }
        // Zero K padding 300..319 (5 x v4bf per row)
        for (int j = lane; j < 16 * 5; j += 32) {
            int r  = j / 5;
            int c4 = DD + (j % 5) * 4;
            v4bf z = {};
            *(v4bf*)&As[w][r][c4] = z;
        }
    }
    __syncthreads();
    if (!live) return;

    const int half = lane >> 4;     // 0 or 1
    const int m    = lane & 15;     // output row within tile (A-row owner)

    // Assemble the ten 16x32 bf16 A-fragments per ISA layout. For lane l,
    // elements 0..7 are K = kt*32 + half*8 + e (contiguous 16B) and elements
    // 8..15 are K = kt*32 + half*8 + 16 + (e-8) (contiguous 16B, +32B):
    // exactly two aligned ds_load_b128 per fragment.
    const __bf16* rowp = &As[w][m][0];
    v16bf a[KT];
#pragma unroll
    for (int kt = 0; kt < KT; ++kt) {
        int off = kt * 32 + half * 8;
        v8bf lo = *(const v8bf*)(rowp + off);
        v8bf hi = *(const v8bf*)(rowp + off + 16);
        a[kt] = __builtin_shufflevector(lo, hi,
                                        0, 1, 2, 3, 4, 5, 6, 7,
                                        8, 9, 10, 11, 12, 13, 14, 15);
    }

    float sp[8];
#pragma unroll
    for (int i = 0; i < 8; ++i) sp[i] = 0.0f;
    const float batt = b_att[0];

    for (int nt = 0; nt < NT; ++nt) {
        const int N = nt * 16 + m;      // this lane's output column
        v8f acc = {};
#pragma unroll
        for (int kt = 0; kt < KT; ++kt) {
            v16bf bf = *(const v16bf*)(wb + (((nt * KT) + kt) * 32 + lane) * 16);
            acc = __builtin_amdgcn_wmma_f32_16x16x32_bf16(
                false, a[kt], false, bf, (short)0, acc, false, false);
        }
        // Fused tanh + dot with W_att column; padded arrays -> no divergence,
        // wpad==0 for N>=300 annihilates the padded columns.
        const float bt = bpad[N];
        const float wa = wpad[N];
#pragma unroll
        for (int i = 0; i < 8; ++i) {
            // c element i: row M = i + 8*half, col N
            sp[i] += fast_tanh(acc[i] + bt) * wa;
        }
    }

    // Reduce over the 16 N-lanes within each half (wave32 shuffles).
#pragma unroll
    for (int i = 0; i < 8; ++i) {
        float v = sp[i];
        v += __shfl_xor(v, 1, 32);
        v += __shfl_xor(v, 2, 32);
        v += __shfl_xor(v, 4, 32);
        v += __shfl_xor(v, 8, 32);
        sp[i] = v;
    }
    if (m == 0) {
        // lane 0 holds rows 0..7, lane 16 holds rows 8..15
#pragma unroll
        for (int i = 0; i < 8; ++i) {
            int row = rowTile * 16 + i + 8 * half;
            scores[row] = sp[i] + batt;
        }
    }
}

// ---------------------------------------------------------------------------
// Kernel 2: softmax over the TIME axis, in-place in the attn output region.
// One block per (b, p); element t sits at [(b*T + t)*P + p].
// ---------------------------------------------------------------------------
__global__ __launch_bounds__(256)
void softmax_time_kernel(float* __restrict__ attn) {
    __shared__ float red[256];
    const int g = blockIdx.x;            // 0 .. B*P-1
    const int b = g / PP, p = g % PP;
    const int tid = threadIdx.x;

    float x = (tid < TT) ? attn[(b * TT + tid) * PP + p] : -1e30f;
    red[tid] = x;
    __syncthreads();
    for (int s = 128; s > 0; s >>= 1) {
        if (tid < s) red[tid] = fmaxf(red[tid], red[tid + s]);
        __syncthreads();
    }
    const float mx = red[0];
    __syncthreads();

    float e = (tid < TT) ? __expf(x - mx) : 0.0f;
    red[tid] = e;
    __syncthreads();
    for (int s = 128; s > 0; s >>= 1) {
        if (tid < s) red[tid] += red[tid + s];
        __syncthreads();
    }
    const float sum = red[0];
    if (tid < TT) attn[(b * TT + tid) * PP + p] = e / sum;
}

// ---------------------------------------------------------------------------
// Kernel 3: code_vectors (attn-weighted re-gather, L2 hits) + MLP head.
// One block per (b,t).
// ---------------------------------------------------------------------------
__global__ __launch_bounds__(128)
void head_kernel(const int* __restrict__ idx,
                 const float* __restrict__ nodes,
                 const float* __restrict__ paths,
                 const float* __restrict__ x_float,
                 const float* __restrict__ W_skill, const float* __restrict__ b_skill,
                 const float* __restrict__ W_kc,    const float* __restrict__ b_kc,
                 const float* __restrict__ W_pred,  const float* __restrict__ b_pred,
                 const float* __restrict__ attn,
                 float* __restrict__ res, float* __restrict__ res2,
                 float* __restrict__ kc_out)
{
    __shared__ float aw[PP];
    __shared__ int   i0s[PP], i1s[PP], i2s[PP];
    __shared__ float cv[DD];
    __shared__ float r2s[SKILL];

    const int bt  = blockIdx.x;          // 0 .. B*T-1
    const int tid = threadIdx.x;

    for (int p = tid; p < PP; p += 128) {
        aw[p] = attn[bt * PP + p];
        int base = (bt * PP + p) * 3;
        i0s[p] = idx[base + 0];
        i1s[p] = idx[base + 1];
        i2s[p] = idx[base + 2];
    }
    __syncthreads();

    for (int d = tid; d < DD; d += 128) {
        float acc = 0.0f;
        if (d < EMB) {
            for (int p = 0; p < PP; ++p) acc += nodes[i0s[p] * EMB + d] * aw[p];
        } else if (d < 2 * EMB) {
            int dd = d - EMB;
            for (int p = 0; p < PP; ++p) acc += nodes[i2s[p] * EMB + dd] * aw[p];
        } else {
            int dd = d - 2 * EMB;
            for (int p = 0; p < PP; ++p) acc += paths[i1s[p] * EMB + dd] * aw[p];
        }
        cv[d] = acc;
    }
    __syncthreads();

    if (tid < SKILL) {
        float o = b_skill[tid];
        for (int d = 0; d < DD; ++d) o += cv[d] * W_skill[d * SKILL + tid];
        float k = b_kc[tid];
        for (int q = 0; q < QQ; ++q) k += x_float[bt * QQ + q] * W_kc[q * SKILL + tid];
        k = 1.0f / (1.0f + __expf(-k));
        float r2 = (k >= 0.5f) ? o : 0.0f;
        r2s[tid] = r2;
        res2[bt * SKILL + tid]   = r2;
        kc_out[bt * SKILL + tid] = k;
    }
    __syncthreads();

    if (tid < OUT) {
        float v = b_pred[tid];
        for (int s = 0; s < SKILL; ++s) v += r2s[s] * W_pred[s * OUT + tid];
        res[bt * OUT + tid] = 1.0f / (1.0f + __expf(-v));
    }
}

// ---------------------------------------------------------------------------
extern "C" void kernel_launch(void* const* d_in, const int* in_sizes, int n_in,
                              void* d_out, int out_size, void* d_ws, size_t ws_size,
                              hipStream_t stream) {
    const float* x_float = (const float*)d_in[0];
    const int*   c2v     = (const int*)d_in[1];
    const float* nodes   = (const float*)d_in[2];
    const float* paths   = (const float*)d_in[3];
    const float* W_trans = (const float*)d_in[4];
    const float* b_trans = (const float*)d_in[5];
    const float* W_att   = (const float*)d_in[6];
    const float* b_att   = (const float*)d_in[7];
    const float* W_skill = (const float*)d_in[8];
    const float* b_skill = (const float*)d_in[9];
    const float* W_kc    = (const float*)d_in[10];
    const float* b_kc    = (const float*)d_in[11];
    const float* W_pred  = (const float*)d_in[12];
    const float* b_pred  = (const float*)d_in[13];

    // d_out layout (flat, return order): res[16*200*50], res2[16*200*30],
    // kc_mask[16*200*30], attn[16*200*100]
    float* out  = (float*)d_out;
    float* res  = out;
    float* res2 = out + BB * TT * OUT;                       // +160000
    float* kc   = out + BB * TT * (OUT + SKILL);             // +256000
    float* attn = out + BB * TT * (OUT + 2 * SKILL);         // +352000

    // ws layout: wb fragments (194,560 B), then padded b_trans / W_att
    __bf16* wb   = (__bf16*)d_ws;
    float*  bpad = (float*)d_ws + WB_FLOATS;
    float*  wpad = bpad + NPAD;

    // 0) pack W_trans into WMMA B-fragment layout + padded bias/att vectors
    pack_wtrans_kernel<<<(WB_ELEMS + 255) / 256, 256, 0, stream>>>(
        W_trans, b_trans, W_att, wb, bpad, wpad);
    // 1) fused transform+attention scores -> attn slot (pre-softmax)
    {
        int blocks = (ROWTILES + WAVES - 1) / WAVES;   // 5000
        score_kernel<<<blocks, WAVES * 32, 0, stream>>>(
            c2v, nodes, paths, wb, bpad, wpad, b_att, attn);
    }
    // 2) softmax over time, in place
    softmax_time_kernel<<<BB * PP, 256, 0, stream>>>(attn);
    // 3) weighted gather + head
    head_kernel<<<BB * TT, 128, 0, stream>>>(
        c2v, nodes, paths, x_float, W_skill, b_skill, W_kc, b_kc,
        W_pred, b_pred, attn, res, res2, kc);
}
